// SwiGLUMoEDownProj_33767032882011
// MI455X (gfx1250) — matrix operand order repaired and checked by reading, not compiled, not run on hardware
//
#include <hip/hip_runtime.h>
#include <hip/hip_bf16.h>

// Problem constants (from reference): B=4, N=2048, D=1024, E=8, R=256, K=2
#define TOK    8192      // B*N tokens
#define DDIM   1024
#define NEXP   8
#define RDIM   256
#define TILE_M 32
#define TILES_PER_E (TOK / TILE_M)   // 256

typedef __attribute__((ext_vector_type(16))) __bf16 bf16x16;
typedef __attribute__((ext_vector_type(8)))  __bf16 bf16x8;
typedef __attribute__((ext_vector_type(4)))  __bf16 bf16x4;
typedef __attribute__((ext_vector_type(8)))  float  f32x8;

static __device__ __forceinline__ bf16x16 ld_frag(const __bf16* p0, const __bf16* p1) {
  bf16x8 lo = *(const bf16x8*)p0;     // global_load_b128 / ds_load_b128
  bf16x8 hi = *(const bf16x8*)p1;
  return __builtin_shufflevector(lo, hi, 0,1,2,3,4,5,6,7,8,9,10,11,12,13,14,15);
}

static __device__ __forceinline__ f32x8 wmma_bf16(bf16x16 a, bf16x16 b, f32x8 c) {
  // D = A(16x32) * B(32x16) + C, f32 accumulate
  return __builtin_amdgcn_wmma_f32_16x16x32_bf16(false, a, false, b, (short)0, c, false, false);
}

// ---------------------------------------------------------------------------
// Kernel 0: zero output (poisoned by harness) and per-expert counters.
// ---------------------------------------------------------------------------
__global__ void zero_out_kernel(float* __restrict__ out, int* __restrict__ cnt, int n4) {
  int i = blockIdx.x * blockDim.x + threadIdx.x;
  if (i < n4) ((float4*)out)[i] = make_float4(0.f, 0.f, 0.f, 0.f);
  if (blockIdx.x == 0 && threadIdx.x < NEXP) cnt[threadIdx.x] = 0;
}

// ---------------------------------------------------------------------------
// Kernel 1: x (f32, [TOK][D]) -> xb (bf16, same layout)
// ---------------------------------------------------------------------------
__global__ void cvt_x_kernel(const float* __restrict__ x, __bf16* __restrict__ xb) {
  int i = blockIdx.x * blockDim.x + threadIdx.x;   // over TOK*DDIM/4
  float4 v = ((const float4*)x)[i];
  bf16x4 o;
  o[0] = (__bf16)v.x; o[1] = (__bf16)v.y; o[2] = (__bf16)v.z; o[3] = (__bf16)v.w;
  ((bf16x4*)xb)[i] = o;
}

// ---------------------------------------------------------------------------
// Kernel 2: Wu/Wv (f32 [E][D][R]) -> bf16 transposed [E][R][D] so each
// WMMA B-column is contiguous along K=D.
// ---------------------------------------------------------------------------
__global__ void cvt_wuv_kernel(const float* __restrict__ Wu, const float* __restrict__ Wv,
                               __bf16* __restrict__ WuT, __bf16* __restrict__ WvT) {
  int i = blockIdx.x * blockDim.x + threadIdx.x;   // over E*D*R
  int e = i / (DDIM * RDIM);
  int rem = i % (DDIM * RDIM);
  int d = rem / RDIM;
  int r = rem % RDIM;
  size_t o = ((size_t)e * RDIM + r) * DDIM + d;
  WuT[o] = (__bf16)Wu[i];
  WvT[o] = (__bf16)Wv[i];
}

// Wo (f32 [E][R][D]) -> bf16 transposed [E][D][R]; B-column contiguous along K=R.
__global__ void cvt_wo_kernel(const float* __restrict__ Wo, __bf16* __restrict__ WoT) {
  int i = blockIdx.x * blockDim.x + threadIdx.x;   // over E*R*D
  int e = i / (RDIM * DDIM);
  int rem = i % (RDIM * DDIM);
  int r = rem / DDIM;
  int d = rem % DDIM;
  WoT[((size_t)e * DDIM + d) * RDIM + r] = (__bf16)Wo[i];
}

// ---------------------------------------------------------------------------
// Kernel 3: router. One wave32 per token: logits = x . gate_w, top-2,
// softmax over the 2, compact (token, weight) into per-expert lists.
// ---------------------------------------------------------------------------
__global__ void router_kernel(const float* __restrict__ x, const float* __restrict__ gate_w,
                              int* __restrict__ cnt, int* __restrict__ idxl,
                              float* __restrict__ gwl) {
  const int wv = threadIdx.x >> 5;
  const int lane = threadIdx.x & 31;
  const int token = blockIdx.x * (blockDim.x >> 5) + wv;
  if (token >= TOK) return;

  const float* xr = x + (size_t)token * DDIM;
  float acc[NEXP];
#pragma unroll
  for (int e = 0; e < NEXP; ++e) acc[e] = 0.f;

  for (int d = lane; d < DDIM; d += 32) {
    float xv = xr[d];
    const float* gr = gate_w + (size_t)d * NEXP;   // row of [D][E]
#pragma unroll
    for (int e = 0; e < NEXP; ++e) acc[e] += xv * gr[e];
  }
#pragma unroll
  for (int e = 0; e < NEXP; ++e)
    for (int off = 16; off > 0; off >>= 1)
      acc[e] += __shfl_xor(acc[e], off, 32);

  if (lane == 0) {
    int i0 = 0; float v0 = acc[0];
#pragma unroll
    for (int e = 1; e < NEXP; ++e) if (acc[e] > v0) { v0 = acc[e]; i0 = e; }
    int i1 = (i0 == 0) ? 1 : 0; float v1 = acc[i1];
#pragma unroll
    for (int e = 0; e < NEXP; ++e) if (e != i0 && acc[e] > v1) { v1 = acc[e]; i1 = e; }
    float ex = __expf(v1 - v0);           // softmax over the two kept logits
    float w0 = 1.f / (1.f + ex);
    float w1 = ex * w0;
    int p0 = atomicAdd(&cnt[i0], 1);
    idxl[i0 * TOK + p0] = token; gwl[i0 * TOK + p0] = w0;
    int p1 = atomicAdd(&cnt[i1], 1);
    idxl[i1 * TOK + p1] = token; gwl[i1 * TOK + p1] = w1;
  }
}

// ---------------------------------------------------------------------------
// Kernel 4: grouped expert GEMM. Block = (expert, 32-token tile), 8 waves.
// Gathered X tile staged in LDS via async-to-LDS DMA; double-buffered
// fragment pipeline feeding v_wmma_f32_16x16x32_bf16.
// ---------------------------------------------------------------------------
__global__ __launch_bounds__(256) void moe_kernel(
    const __bf16* __restrict__ xb,
    const __bf16* __restrict__ WuT, const __bf16* __restrict__ WvT,
    const __bf16* __restrict__ WoT,
    const int* __restrict__ cnt, const int* __restrict__ idxl,
    const float* __restrict__ gwl, float* __restrict__ out) {
  __shared__ __bf16 sX[TILE_M][DDIM];   // 64 KB gathered activations (bf16)
  __shared__ __bf16 sS[TILE_M][RDIM];   // 16 KB silu(u)*v staged for GEMM2
  __shared__ int   sTok[TILE_M];
  __shared__ float sGw[TILE_M];

  const int e = blockIdx.x / TILES_PER_E;
  const int tile = blockIdx.x % TILES_PER_E;
  const int tstart = tile * TILE_M;
  const int ce = cnt[e];
  if (tstart >= ce) return;            // block-uniform early exit

  if (threadIdx.x < TILE_M) {
    int g = tstart + threadIdx.x;
    if (g < ce) { sTok[threadIdx.x] = idxl[e * TOK + g]; sGw[threadIdx.x] = gwl[e * TOK + g]; }
    else        { sTok[threadIdx.x] = 0;                 sGw[threadIdx.x] = 0.f; }  // padded rows: weight 0
  }
  __syncthreads();

  // ---- Async gather of the 32 token rows into LDS (64 KB, b128 granules).
  // 256 threads: thread t copies 128 columns of row t>>3 (16 x b128).
  {
    const int row = threadIdx.x >> 3;
    const int cseg = (threadIdx.x & 7) * 128;
    const __bf16* src = xb + (size_t)sTok[row] * DDIM + cseg;
    unsigned dst = (unsigned)(unsigned long long)&sX[row][cseg];  // LDS byte offset
#pragma unroll
    for (int c = 0; c < 16; ++c) {
      asm volatile("global_load_async_to_lds_b128 %0, %1, off"
                   :: "v"(dst + (unsigned)(c * 16)), "v"(src + c * 8) : "memory");
    }
    asm volatile("s_wait_asynccnt 0x0" ::: "memory");
  }
  __syncthreads();

  const int lane = threadIdx.x & 31;
  const int wv = threadIdx.x >> 5;
  const int r  = lane & 15;            // A: row-in-tile / B,C,D: column-in-tile
  const int hf = lane >> 4;            // K-half selector
  const int rt = wv & 1;               // which 16-row tile of the 32
  const int cg = wv >> 1;              // 0..3

  const __bf16* aLds = &sX[rt * 16 + r][0];

  // ---- GEMM 1: u and v, double-buffered fragment pipeline over K=1024.
  const __bf16* bu_base[4];
  const __bf16* bv_base[4];
#pragma unroll
  for (int j = 0; j < 4; ++j) {
    const int col = (cg * 4 + j) * 16 + r;
    bu_base[j] = WuT + ((size_t)e * RDIM + col) * DDIM + hf * 16;
    bv_base[j] = WvT + ((size_t)e * RDIM + col) * DDIM + hf * 16;
  }

  f32x8 accU[4] = {};
  f32x8 accV[4] = {};
  bf16x16 aS[2];
  bf16x16 buS[2][4], bvS[2][4];

  auto stage1 = [&](int s, int kb) {
    aS[s] = ld_frag(aLds + kb + hf * 16, aLds + kb + hf * 16 + 8);   // ds_load_b128
#pragma unroll
    for (int j = 0; j < 4; ++j) {
      buS[s][j] = ld_frag(bu_base[j] + kb, bu_base[j] + kb + 8);    // global_load_b128
      bvS[s][j] = ld_frag(bv_base[j] + kb, bv_base[j] + kb + 8);
    }
  };
  auto compute1 = [&](int s) {
#pragma unroll
    for (int j = 0; j < 4; ++j) {
      accU[j] = wmma_bf16(aS[s], buS[s][j], accU[j]);
      accV[j] = wmma_bf16(aS[s], bvS[s][j], accV[j]);
    }
  };

  stage1(0, 0);
#pragma unroll 1
  for (int kb = 0; kb < DDIM; kb += 64) {
    stage1(1, kb + 32);                 // prefetch next stage while computing
    compute1(0);
    if (kb + 64 < DDIM) stage1(0, kb + 64);
    compute1(1);
  }

  // ---- s = silu(u) * v  -> LDS (bf16), C-layout: lane = col, vgpr i = row hf*8+i
#pragma unroll
  for (int j = 0; j < 4; ++j) {
#pragma unroll
    for (int i = 0; i < 8; ++i) {
      float u = accU[j][i], v = accV[j][i];
      float s = (u / (1.f + __expf(-u))) * v;
      sS[rt * 16 + hf * 8 + i][(cg * 4 + j) * 16 + r] = (__bf16)s;
    }
  }
  __syncthreads();

  // ---- GEMM 2: y = s * Wo (K=256); 64 col tiles; wave owns 16 in 4 chunks of 4
  const __bf16* srow = &sS[rt * 16 + r][0];
#pragma unroll 1
  for (int c4 = 0; c4 < 4; ++c4) {
    const __bf16* bo_base[4];
#pragma unroll
    for (int j = 0; j < 4; ++j) {
      const int col = (cg * 16 + c4 * 4 + j) * 16 + r;
      bo_base[j] = WoT + ((size_t)e * DDIM + col) * RDIM + hf * 16;
    }

    f32x8 accY[4] = {};
    bf16x16 aS2[2];
    bf16x16 boS[2][4];

    auto stage2 = [&](int s, int kb) {
      aS2[s] = ld_frag(srow + kb + hf * 16, srow + kb + hf * 16 + 8);
#pragma unroll
      for (int j = 0; j < 4; ++j)
        boS[s][j] = ld_frag(bo_base[j] + kb, bo_base[j] + kb + 8);
    };
    auto compute2 = [&](int s) {
#pragma unroll
      for (int j = 0; j < 4; ++j)
        accY[j] = wmma_bf16(aS2[s], boS[s][j], accY[j]);
    };

    stage2(0, 0);
#pragma unroll 1
    for (int kb = 0; kb < RDIM; kb += 64) {
      stage2(1, kb + 32);
      compute2(0);
      if (kb + 64 < RDIM) stage2(0, kb + 64);
      compute2(1);
    }

#pragma unroll
    for (int j = 0; j < 4; ++j) {
      const int col = (cg * 16 + c4 * 4 + j) * 16 + r;
#pragma unroll
      for (int i = 0; i < 8; ++i) {
        const int m = rt * 16 + hf * 8 + i;
        atomicAdd(out + (size_t)sTok[m] * DDIM + col, sGw[m] * accY[j][i]);
      }
    }
  }
}

// ---------------------------------------------------------------------------
extern "C" void kernel_launch(void* const* d_in, const int* in_sizes, int n_in,
                              void* d_out, int out_size, void* d_ws, size_t ws_size,
                              hipStream_t stream) {
  const float* x      = (const float*)d_in[0];   // [4,2048,1024]
  const float* gate_w = (const float*)d_in[1];   // [1024,8]
  const float* Wu     = (const float*)d_in[2];   // [8,1024,256]
  const float* Wv     = (const float*)d_in[3];   // [8,1024,256]
  const float* Wo     = (const float*)d_in[4];   // [8,256,1024]
  float* out = (float*)d_out;                    // [4,2048,1024]

  char* ws = (char*)d_ws;
  size_t off = 0;
  __bf16* xb  = (__bf16*)(ws + off); off += (size_t)TOK * DDIM * 2;          // 16 MB
  __bf16* WuT = (__bf16*)(ws + off); off += (size_t)NEXP * DDIM * RDIM * 2;  //  4 MB
  __bf16* WvT = (__bf16*)(ws + off); off += (size_t)NEXP * DDIM * RDIM * 2;  //  4 MB
  __bf16* WoT = (__bf16*)(ws + off); off += (size_t)NEXP * DDIM * RDIM * 2;  //  4 MB
  int*   cnt  = (int*)(ws + off);    off += 256;
  int*   idxl = (int*)(ws + off);    off += (size_t)NEXP * TOK * 4;          // 256 KB
  float* gwl  = (float*)(ws + off);  off += (size_t)NEXP * TOK * 4;          // 256 KB
  (void)ws_size; (void)in_sizes; (void)n_in; (void)out_size;

  const int n4 = TOK * DDIM / 4;
  zero_out_kernel<<<(n4 + 255) / 256, 256, 0, stream>>>(out, cnt, n4);
  cvt_x_kernel<<<n4 / 256, 256, 0, stream>>>(x, xb);
  cvt_wuv_kernel<<<(NEXP * DDIM * RDIM) / 256, 256, 0, stream>>>(Wu, Wv, WuT, WvT);
  cvt_wo_kernel<<<(NEXP * RDIM * DDIM) / 256, 256, 0, stream>>>(Wo, WoT);
  router_kernel<<<TOK / 8, 256, 0, stream>>>(x, gate_w, cnt, idxl, gwl);
  moe_kernel<<<NEXP * TILES_PER_E, 256, 0, stream>>>(xb, WuT, WvT, WoT, cnt, idxl, gwl, out);
}
